// Attn_32942399160704
// MI455X (gfx1250) — matrix operand order repaired and checked by reading, not compiled, not run on hardware
//
#include <hip/hip_runtime.h>

// Problem constants (B=16, T=2048, H=1024)
#define B_SZ 16
#define T_SZ 2048
#define H_SZ 1024
#define K_SZ 2048          // 2H, GEMM K
#define W_STRIDE 3072      // 3H row stride of W
#define KBLK 64            // K block per LDS buffer
#define BS_STRIDE 72       // halves per LDS row: 64 + 8 pad -> 144B (16B aligned, bank-conflict-free)

typedef __attribute__((ext_vector_type(16))) __bf16 v16bf;
typedef __attribute__((ext_vector_type(8)))  float  v8f;

// pack two f32 -> one dword of two bf16.
// Preferred: HW packed convert (if toolchain exposes it). Otherwise a single
// v_perm_b32 byte-select of the two high halves (truncation; fine at bf16).
#if __has_builtin(__builtin_amdgcn_cvt_pk_bf16_f32)
__device__ __forceinline__ unsigned int pk2(float a, float b) {
  auto r = __builtin_amdgcn_cvt_pk_bf16_f32(a, b);
  return __builtin_bit_cast(unsigned int, r);
}
#else
__device__ __forceinline__ unsigned int pk2(float a, float b) {
  unsigned int ua = __builtin_bit_cast(unsigned int, a);
  unsigned int ub = __builtin_bit_cast(unsigned int, b);
  // result bytes (LE): {ua[2], ua[3], ub[2], ub[3]} ; pool: bytes0-3=src1(ua), 4-7=src0(ub)
  return __builtin_amdgcn_perm(ub, ua, 0x07060302u);
}
#endif

union FragU {
  unsigned int u32[8];
  uint4 q[2];
  v16bf v;
};

// ---- async global->LDS copy (CDNA5), tracked by ASYNCcnt -------------------
__device__ __forceinline__ void async_cp16(const unsigned short* g, unsigned lds_byte_off) {
  asm volatile("global_load_async_to_lds_b128 %0, %1, off"
               :: "v"(lds_byte_off), "v"((unsigned long long)(size_t)g)
               : "memory");
}
__device__ __forceinline__ void wait_async0() {
  asm volatile("s_wait_asynccnt 0x0" ::: "memory");
}

// hb[b][h] = bias[h] + hidden[b,:] . W[h, 0:H]   (t-invariant part of the contraction)
__global__ void __launch_bounds__(256)
hb_kernel(const float* __restrict__ hidden, const float* __restrict__ W,
          const float* __restrict__ bias, float* __restrict__ hb) {
  int idx = blockIdx.x * blockDim.x + threadIdx.x;   // [0, B*H)
  int b = idx >> 10, h = idx & (H_SZ - 1);
  const float4* hp = (const float4*)(hidden + b * H_SZ);
  const float4* wp = (const float4*)(W + (size_t)h * W_STRIDE);
  float s = bias[h];
#pragma unroll 4
  for (int k = 0; k < H_SZ / 4; ++k) {
    float4 a = hp[k], w = wp[k];
    s += a.x * w.x + a.y * w.y + a.z * w.z + a.w * w.w;
  }
  hb[idx] = s;
}

// One-shot W2 -> bf16 conversion: Wb[n][k] = bf16(W[n][H+k]), row-major (1024 x 2048)
__global__ void __launch_bounds__(256)
wconv_kernel(const float* __restrict__ W, unsigned short* __restrict__ Wb) {
  int idx = blockIdx.x * blockDim.x + threadIdx.x;   // each thread: 8 elements
  int n = idx >> 8;                 // 256 8-chunks per row
  int k = (idx & 255) * 8;
  const float4* src = (const float4*)(W + (size_t)n * W_STRIDE + H_SZ + k);
  float4 f0 = src[0], f1 = src[1];
  uint4 p;
  p.x = pk2(f0.x, f0.y); p.y = pk2(f0.z, f0.w);
  p.z = pk2(f1.x, f1.y); p.w = pk2(f1.z, f1.w);
  *(uint4*)(Wb + (size_t)n * K_SZ + k) = p;
}

// Fused GEMM (enc @ W2^T) + tanh + v-dot.  64 M-rows per workgroup (4 waves),
// N in chunks of 128 (8 C tiles/wave), B tile async-DMA'd to double-buffered LDS.
__global__ void __launch_bounds__(128)
attn_gemm_kernel(const float* __restrict__ enc,          // (B*T, 2H) row-major
                 const unsigned short* __restrict__ Wb,  // (H, 2H) bf16 row-major
                 const float* __restrict__ hb,           // (B, H)
                 const float* __restrict__ vvec,         // (H)
                 float* __restrict__ scores)             // (B, T)
{
  __shared__ unsigned short Bs[2][128 * BS_STRIDE];  // 2 x 18432 B

  const int tid  = threadIdx.x;
  const int lane = tid & 31;
  const int wave = tid >> 5;        // 0..3
  const int hlf  = lane >> 4;       // k-group select (ISA 16-bit A/B layout)
  const int ncol = lane & 15;

  const int mblock = blockIdx.x;    // 0..511
  const int b      = mblock >> 5;   // 32 blocks of 64 rows per batch
  const int m0     = mblock * 64 + wave * 16;
  const float* Arow = enc + (size_t)(m0 + ncol) * K_SZ;

  float part[8];
#pragma unroll
  for (int r = 0; r < 8; ++r) part[r] = 0.f;

  const unsigned lds0 = (unsigned)(size_t)&Bs[0][0];
  const unsigned lds1 = (unsigned)(size_t)&Bs[1][0];

  // Fill Bs[buf][n][0..63] = Wb[n0+n][kb .. kb+64) via per-lane async 16B DMA.
  // 128 rows * 8 chunks = 1024 chunks; thread covers chunks q*128+tid, q=0..7.
  auto issue_fill = [&](unsigned ldsbase, int n0, int kb) {
#pragma unroll
    for (int q = 0; q < 8; ++q) {
      int c   = q * 128 + tid;
      int row = c >> 3;
      int jk  = c & 7;
      async_cp16(Wb + (size_t)(n0 + row) * K_SZ + kb + jk * 8,
                 ldsbase + row * (BS_STRIDE * 2) + jk * 16);
    }
  };

  const int NFILL = K_SZ / KBLK;    // 32

  for (int nc = 0; nc < H_SZ / 128; ++nc) {       // 8 N-chunks
    const int n0 = nc * 128;
    v8f c[8] = {};

    issue_fill(lds0, n0, 0);        // prologue fill of buffer 0
    wait_async0();
    __syncthreads();

    for (int kbi = 0; kbi < NFILL; ++kbi) {
      const unsigned short* cur = (kbi & 1) ? &Bs[1][0] : &Bs[0][0];
      if (kbi + 1 < NFILL)          // overlap DMA of next K block with compute
        issue_fill((kbi & 1) ? lds0 : lds1, n0, (kbi + 1) * KBLK);

      const int kb = kbi * KBLK;
#pragma unroll
      for (int ks = 0; ks < KBLK / 32; ++ks) {
        const int kk = kb + ks * 32;
        // A fragment: 16-bit A 16x32 layout — lanes<16: k {kk..+7, kk+16..+23},
        // lanes>=16 shifted by 8.  8 contiguous f32 per half-group, cvt in-register.
        FragU a;
        {
          const int klo = kk + hlf * 8;
          float4 f0 = *(const float4*)(Arow + klo);
          float4 f1 = *(const float4*)(Arow + klo + 4);
          float4 f2 = *(const float4*)(Arow + klo + 16);
          float4 f3 = *(const float4*)(Arow + klo + 20);
          a.u32[0] = pk2(f0.x, f0.y); a.u32[1] = pk2(f0.z, f0.w);
          a.u32[2] = pk2(f1.x, f1.y); a.u32[3] = pk2(f1.z, f1.w);
          a.u32[4] = pk2(f2.x, f2.y); a.u32[5] = pk2(f2.z, f2.w);
          a.u32[6] = pk2(f3.x, f3.y); a.u32[7] = pk2(f3.z, f3.w);
        }
        // preload all 8 B fragments so ds waits drain progressively
        FragU bf[8];
#pragma unroll
        for (int j = 0; j < 8; ++j) {
          const unsigned short* bp = cur + (j * 16 + ncol) * BS_STRIDE + ks * 32 + hlf * 8;
          bf[j].q[0] = *(const uint4*)(bp);
          bf[j].q[1] = *(const uint4*)(bp + 16);
        }
#pragma unroll
        for (int j = 0; j < 8; ++j)
          c[j] = __builtin_amdgcn_wmma_f32_16x16x32_bf16(
              false, a.v, false, bf[j].v, (short)0, c[j], false, false);
      }

      wait_async0();                // own DMA done before publishing at barrier
      __syncthreads();
    }

    // fused epilogue: part[r] += tanh(E + hb[n]) * v[n], per this lane's columns
#pragma unroll
    for (int j = 0; j < 8; ++j) {
      const int n    = n0 + j * 16 + ncol;
      const float h  = hb[b * H_SZ + n];
      const float vv = vvec[n];
#pragma unroll
      for (int r = 0; r < 8; ++r)
        part[r] += tanhf(c[j][r] + h) * vv;
    }
  }

  // reduce across the 16 lanes holding each row group (C layout: rows r+8*hlf)
#pragma unroll
  for (int r = 0; r < 8; ++r) {
    float s = part[r];
    s += __shfl_xor(s, 1, 32);
    s += __shfl_xor(s, 2, 32);
    s += __shfl_xor(s, 4, 32);
    s += __shfl_xor(s, 8, 32);
    part[r] = s;
  }
  if (ncol == 0) {
    const int tbase = (mblock & 31) * 64 + wave * 16 + hlf * 8;
#pragma unroll
    for (int r = 0; r < 8; ++r)
      scores[b * T_SZ + tbase + r] = part[r];
  }
}

// Softmax over T per batch row
__global__ void __launch_bounds__(256)
softmax_kernel(const float* __restrict__ scores, float* __restrict__ out) {
  __shared__ float red[256];
  const int b = blockIdx.x, tid = threadIdx.x;
  const float* row = scores + b * T_SZ;
  float vals[8];
  float vmax = -3.402823466e38f;
#pragma unroll
  for (int i = 0; i < 8; ++i) { vals[i] = row[tid + i * 256]; vmax = fmaxf(vmax, vals[i]); }
  red[tid] = vmax; __syncthreads();
  for (int s = 128; s > 0; s >>= 1) { if (tid < s) red[tid] = fmaxf(red[tid], red[tid + s]); __syncthreads(); }
  const float m = red[0]; __syncthreads();
  float sum = 0.f;
#pragma unroll
  for (int i = 0; i < 8; ++i) { vals[i] = expf(vals[i] - m); sum += vals[i]; }
  red[tid] = sum; __syncthreads();
  for (int s = 128; s > 0; s >>= 1) { if (tid < s) red[tid] += red[tid + s]; __syncthreads(); }
  const float inv = 1.f / red[0];
#pragma unroll
  for (int i = 0; i < 8; ++i) out[b * T_SZ + tid + i * 256] = vals[i] * inv;
}

extern "C" void kernel_launch(void* const* d_in, const int* in_sizes, int n_in,
                              void* d_out, int out_size, void* d_ws, size_t ws_size,
                              hipStream_t stream) {
  const float* hidden = (const float*)d_in[0];
  const float* enc    = (const float*)d_in[1];
  const float* W      = (const float*)d_in[2];
  const float* bias   = (const float*)d_in[3];
  const float* vvec   = (const float*)d_in[4];
  float* out = (float*)d_out;

  float* hb              = (float*)d_ws;                     // B*H f32   (64 KB)
  float* scores          = hb + B_SZ * H_SZ;                 // B*T f32   (128 KB)
  unsigned short* Wb     = (unsigned short*)(scores + B_SZ * T_SZ);  // H*2H bf16 (4 MB)

  hb_kernel<<<(B_SZ * H_SZ) / 256, 256, 0, stream>>>(hidden, W, bias, hb);
  wconv_kernel<<<(H_SZ * K_SZ / 8) / 256, 256, 0, stream>>>(W, Wb);
  attn_gemm_kernel<<<(B_SZ * T_SZ) / 64, 128, 0, stream>>>(enc, Wb, hb, vvec, scores);
  softmax_kernel<<<B_SZ, 256, 0, stream>>>(scores, out);
}
